// GATModel_5214090297617
// MI455X (gfx1250) — matrix operand order repaired and checked by reading, not compiled, not run on hardware
//
#include <hip/hip_runtime.h>
#include <math.h>

typedef __attribute__((ext_vector_type(16))) _Float16 v16h;
typedef __attribute__((ext_vector_type(8)))  float    v8f;

#define N_NODES 100000
#define N_EDGES 1600000
#define E_TOT   (N_EDGES + N_NODES)
#define NEG_SLOPE 0.2f
#define ROWS_PER_WAVE 10

__device__ __forceinline__ float leaky(float x) { return x > 0.f ? x : NEG_SLOPE * x; }

// Monotonic float<->uint encoding so segment-max can use native uint atomicMax.
__device__ __forceinline__ unsigned f2ord(float f) {
    unsigned u = __float_as_uint(f);
    return (u & 0x80000000u) ? ~u : (u | 0x80000000u);
}
__device__ __forceinline__ float ord2f(unsigned u) {
    unsigned v = (u & 0x80000000u) ? (u & 0x7FFFFFFFu) : ~u;
    return __uint_as_float(v);
}

// ---------------------------------------------------------------------------
// WMMA GEMM: out[M x Ncols] = X[M x 128] * W[128 x Ncols], f32 data, f16 MACs.
// One wave32 per 16-wide column strip; B fragments (full K=128) preloaded once
// and reused across ROWS_PER_WAVE row tiles. A loaded as float4 (b128) per the
// contiguous 8-element K-runs of the CDNA5 16-bit A layout (ISA 7.12.2).
// ---------------------------------------------------------------------------
__global__ void gemm_wmma(const float* __restrict__ X, const float* __restrict__ W,
                          float* __restrict__ out, int M, int Ncols)
{
    const int K = 128;
    int wave = (int)((blockIdx.x * blockDim.x + threadIdx.x) >> 5);
    int lane = threadIdx.x & 31;
    int colTiles = Ncols >> 4;
    int strip   = wave / colTiles;
    int colTile = wave % colTiles;
    int rowBase0 = strip * ROWS_PER_WAVE * 16;
    if (rowBase0 >= M) return;
    int colBase = colTile * 16;
    int half16 = lane >> 4;   // which K-half this lane holds
    int lane16 = lane & 15;

    // Preload B fragments for all 4 K-steps (reused for every row tile).
    // B (32x16 f16): elem e -> K = e + 8*((e>=8)+half) (N=lane%16), matching A.
    v16h b[4];
    #pragma unroll
    for (int kt = 0; kt < 4; ++kt) {
        #pragma unroll
        for (int e = 0; e < 16; ++e) {
            int kb = kt * 32 + half16 * 8 + e + ((e >= 8) ? 8 : 0);
            b[kt][e] = (_Float16)W[(size_t)kb * Ncols + colBase + lane16];
        }
    }

    for (int r = 0; r < ROWS_PER_WAVE; ++r) {
        int rowBase = rowBase0 + r * 16;
        if (rowBase >= M) break;
        const float* xrow = X + (size_t)(rowBase + lane16) * K;
        v8f c = {};
        #pragma unroll
        for (int kt = 0; kt < 4; ++kt) {
            // A (16x32 f16): lane = M%16 (+16 selects K-half);
            // elem e -> K = e + 8*((e>=8)+half): two contiguous 8-float runs.
            const float4 r0 = *(const float4*)(xrow + kt * 32 + 8 * half16);
            const float4 r1 = *(const float4*)(xrow + kt * 32 + 8 * half16 + 4);
            const float4 r2 = *(const float4*)(xrow + kt * 32 + 16 + 8 * half16);
            const float4 r3 = *(const float4*)(xrow + kt * 32 + 16 + 8 * half16 + 4);
            v16h a;
            a[0]  = (_Float16)r0.x; a[1]  = (_Float16)r0.y;
            a[2]  = (_Float16)r0.z; a[3]  = (_Float16)r0.w;
            a[4]  = (_Float16)r1.x; a[5]  = (_Float16)r1.y;
            a[6]  = (_Float16)r1.z; a[7]  = (_Float16)r1.w;
            a[8]  = (_Float16)r2.x; a[9]  = (_Float16)r2.y;
            a[10] = (_Float16)r2.z; a[11] = (_Float16)r2.w;
            a[12] = (_Float16)r3.x; a[13] = (_Float16)r3.y;
            a[14] = (_Float16)r3.z; a[15] = (_Float16)r3.w;
            c = __builtin_amdgcn_wmma_f32_16x16x32_f16(false, a, false, b[kt], (short)0, c,
                                                       false, false);
        }
        // C/D: VGPR rr -> M = rr + 8*(lane>=16), N = lane%16
        #pragma unroll
        for (int rr = 0; rr < 8; ++rr) {
            out[(size_t)(rowBase + rr + 8 * half16) * Ncols + colBase + lane16] = c[rr];
        }
    }
}

// ---------------------------------------------------------------------------
// Zero accumulators / denominators, set segment-max buffers to -inf (encoded).
// ---------------------------------------------------------------------------
__global__ void init_ws(float* acc1, float* denom1, unsigned* emax1,
                        float* acc2, float* denom2, unsigned* emax2)
{
    int i = blockIdx.x * blockDim.x + threadIdx.x;
    if (i < N_NODES * 128) acc1[i] = 0.f;
    if (i < N_NODES * 16)  acc2[i] = 0.f;
    if (i < N_NODES * 4) { denom1[i] = 0.f; emax1[i] = f2ord(-INFINITY); }
    if (i < N_NODES)     { denom2[i] = 0.f; emax2[i] = f2ord(-INFINITY); }
}

// Per-node attention terms: al[n,h] = <xw[n,h,:], a_src[h,:]>, ar likewise.
__global__ void att_terms(const float* __restrict__ xw, const float* __restrict__ a_src,
                          const float* __restrict__ a_dst, float* __restrict__ al,
                          float* __restrict__ ar, int H, int C)
{
    int t = blockIdx.x * blockDim.x + threadIdx.x;   // n*H + h
    if (t >= N_NODES * H) return;
    int h = t % H;
    int n = t / H;
    const float* v = xw + (size_t)n * H * C + (size_t)h * C;
    float s = 0.f, d = 0.f;
    for (int c = 0; c < C; ++c) { s += v[c] * a_src[h * C + c]; d += v[c] * a_dst[h * C + c]; }
    al[t] = s; ar[t] = d;
}

// Per-edge segment max (ordered-uint atomicMax).
__global__ void edge_max(const int* __restrict__ ei, const float* __restrict__ al,
                         const float* __restrict__ ar, unsigned* __restrict__ emax, int H)
{
    long long e = (long long)blockIdx.x * blockDim.x + threadIdx.x;
    if (e >= E_TOT) return;
    int s, d;
    if (e < N_EDGES) { s = ei[e]; d = ei[N_EDGES + e]; }
    else             { s = d = (int)(e - N_EDGES); }      // self loop
    for (int h = 0; h < H; ++h) {
        float v = leaky(al[s * H + h] + ar[d * H + h]);
        atomicMax(&emax[d * H + h], f2ord(v));
    }
}

// Layer 1 scatter: one wave32 per edge; lane owns channels {h*32+lane : h=0..3}.
// Accumulates denom[dst,h] += e^m and acc[dst,ch] += e^m * xw[src,ch].
__global__ void edge_scatter1(const int* __restrict__ ei, const float* __restrict__ al,
                              const float* __restrict__ ar, const unsigned* __restrict__ emax,
                              const float* __restrict__ xw, float* __restrict__ denom,
                              float* __restrict__ acc)
{
    long long wave = ((long long)blockIdx.x * blockDim.x + threadIdx.x) >> 5;
    int lane = threadIdx.x & 31;
    if (wave >= E_TOT) return;
    int s, d;
    if (wave < N_EDGES) { s = ei[wave]; d = ei[N_EDGES + wave]; }
    else                { s = d = (int)(wave - N_EDGES); }
    float em0 = __expf(leaky(al[s * 4 + 0] + ar[d * 4 + 0]) - ord2f(emax[d * 4 + 0]));
    float em1 = __expf(leaky(al[s * 4 + 1] + ar[d * 4 + 1]) - ord2f(emax[d * 4 + 1]));
    float em2 = __expf(leaky(al[s * 4 + 2] + ar[d * 4 + 2]) - ord2f(emax[d * 4 + 2]));
    float em3 = __expf(leaky(al[s * 4 + 3] + ar[d * 4 + 3]) - ord2f(emax[d * 4 + 3]));
    if (lane < 4) {
        float dm = (lane == 0) ? em0 : (lane == 1) ? em1 : (lane == 2) ? em2 : em3;
        atomicAdd(&denom[d * 4 + lane], dm);
    }
    atomicAdd(&acc[(size_t)d * 128 +  0 + lane], em0 * xw[(size_t)s * 128 +  0 + lane]);
    atomicAdd(&acc[(size_t)d * 128 + 32 + lane], em1 * xw[(size_t)s * 128 + 32 + lane]);
    atomicAdd(&acc[(size_t)d * 128 + 64 + lane], em2 * xw[(size_t)s * 128 + 64 + lane]);
    atomicAdd(&acc[(size_t)d * 128 + 96 + lane], em3 * xw[(size_t)s * 128 + 96 + lane]);
}

// h = relu(acc/denom + b1)
__global__ void finalize1(const float* __restrict__ acc, const float* __restrict__ denom,
                          const float* __restrict__ b1, float* __restrict__ hout)
{
    int i = blockIdx.x * blockDim.x + threadIdx.x;
    if (i >= N_NODES * 128) return;
    int n = i >> 7, ch = i & 127, h = ch >> 5;
    float v = acc[i] / denom[n * 4 + h] + b1[ch];
    hout[i] = v > 0.f ? v : 0.f;
}

// Layer 2 scatter: 16 threads per edge (one per channel), single head.
__global__ void edge_scatter2(const int* __restrict__ ei, const float* __restrict__ al,
                              const float* __restrict__ ar, const unsigned* __restrict__ emax,
                              const float* __restrict__ xw, float* __restrict__ denom,
                              float* __restrict__ acc)
{
    long long t = (long long)blockIdx.x * blockDim.x + threadIdx.x;
    long long e = t >> 4;
    int c = (int)(t & 15);
    if (e >= E_TOT) return;
    int s, d;
    if (e < N_EDGES) { s = ei[e]; d = ei[N_EDGES + e]; }
    else             { s = d = (int)(e - N_EDGES); }
    float em = __expf(leaky(al[s] + ar[d]) - ord2f(emax[d]));
    if (c == 0) atomicAdd(&denom[d], em);
    atomicAdd(&acc[(size_t)d * 16 + c], em * xw[(size_t)s * 16 + c]);
}

// out = acc2/denom2 + b2  (no relu on final layer)
__global__ void finalize2(const float* __restrict__ acc, const float* __restrict__ denom,
                          const float* __restrict__ b2, float* __restrict__ out)
{
    int i = blockIdx.x * blockDim.x + threadIdx.x;
    if (i >= N_NODES * 16) return;
    int n = i >> 4, c = i & 15;
    out[i] = acc[i] / denom[n] + b2[c];
}

extern "C" void kernel_launch(void* const* d_in, const int* in_sizes, int n_in,
                              void* d_out, int out_size, void* d_ws, size_t ws_size,
                              hipStream_t stream)
{
    (void)in_sizes; (void)n_in; (void)out_size; (void)ws_size;
    const float* x     = (const float*)d_in[0];
    const int*   ei    = (const int*)d_in[1];
    const float* W1    = (const float*)d_in[2];
    const float* asrc1 = (const float*)d_in[3];
    const float* adst1 = (const float*)d_in[4];
    const float* b1    = (const float*)d_in[5];
    const float* W2    = (const float*)d_in[6];
    const float* asrc2 = (const float*)d_in[7];
    const float* adst2 = (const float*)d_in[8];
    const float* b2    = (const float*)d_in[9];
    float* out = (float*)d_out;
    float* ws  = (float*)d_ws;

    // Workspace layout (floats). xw1 buffer is reused for h after finalize1.
    float*    xw1    = ws;                         // 12.8M  (then h)
    float*    acc1   = ws + 12800000;              // 12.8M
    float*    al1    = ws + 25600000;              // 400k
    float*    ar1    = ws + 26000000;              // 400k
    unsigned* emax1  = (unsigned*)(ws + 26400000); // 400k
    float*    denom1 = ws + 26800000;              // 400k
    float*    xw2    = ws + 27200000;              // 1.6M
    float*    acc2   = ws + 28800000;              // 1.6M
    float*    al2    = ws + 30400000;              // 100k
    float*    ar2    = ws + 30500000;              // 100k
    unsigned* emax2  = (unsigned*)(ws + 30600000); // 100k
    float*    denom2 = ws + 30700000;              // 100k

    const int B = 256;
    const int strips = (N_NODES / 16 + ROWS_PER_WAVE - 1) / ROWS_PER_WAVE;  // 625

    // Re-init scratch every launch (graph-replay deterministic).
    init_ws<<<(N_NODES * 128 + B - 1) / B, B, 0, stream>>>(acc1, denom1, emax1,
                                                           acc2, denom2, emax2);
    // Layer 1
    gemm_wmma<<<(strips * 8 * 32 + B - 1) / B, B, 0, stream>>>(x, W1, xw1, N_NODES, 128);
    att_terms<<<(N_NODES * 4 + B - 1) / B, B, 0, stream>>>(xw1, asrc1, adst1, al1, ar1, 4, 32);
    edge_max<<<(E_TOT + B - 1) / B, B, 0, stream>>>(ei, al1, ar1, emax1, 4);
    edge_scatter1<<<(int)(((long long)E_TOT * 32 + B - 1) / B), B, 0, stream>>>(
        ei, al1, ar1, emax1, xw1, denom1, acc1);
    finalize1<<<(N_NODES * 128 + B - 1) / B, B, 0, stream>>>(acc1, denom1, b1, xw1);
    // Layer 2 (h lives in xw1 buffer)
    gemm_wmma<<<(strips * 1 * 32 + B - 1) / B, B, 0, stream>>>(xw1, W2, xw2, N_NODES, 16);
    att_terms<<<(N_NODES + B - 1) / B, B, 0, stream>>>(xw2, asrc2, adst2, al2, ar2, 1, 16);
    edge_max<<<(E_TOT + B - 1) / B, B, 0, stream>>>(ei, al2, ar2, emax2, 1);
    edge_scatter2<<<(int)(((long long)E_TOT * 16 + B - 1) / B), B, 0, stream>>>(
        ei, al2, ar2, emax2, xw2, denom2, acc2);
    finalize2<<<(N_NODES * 16 + B - 1) / B, B, 0, stream>>>(acc2, denom2, b2, out);
}
